// StructureDiffusion_25881472925819
// MI455X (gfx1250) — matrix operand order repaired and checked by reading, not compiled, not run on hardware
//
#include <hip/hip_runtime.h>

// ---------------------------------------------------------------------------
// StructureDiffusion pair-feature + MLP, fused for gfx1250 (MI455X).
// GEMMs on v_wmma_f32_16x16x32_f16 (fp32 accumulate). W1 is DMA'd into LDS
// with GLOBAL_LOAD_ASYNC_TO_LDS_B128 (overlapped with the geometry phase),
// W2 streams from L2/WGP$. Branchless sigmoid-form gelu (x * sigma(2u)).
// ---------------------------------------------------------------------------

#define DEVI __device__ __forceinline__
#define AS1 __attribute__((address_space(1)))
#define AS3 __attribute__((address_space(3)))

#if defined(__AMDGCN__) && __has_builtin(__builtin_amdgcn_global_load_async_to_lds_b128) && __has_builtin(__builtin_amdgcn_s_wait_asynccnt)
#define USE_ASYNC_LDS 1
#else
#define USE_ASYNC_LDS 0
#endif

constexpr int NN = 32768, KN = 32, AN = 4, PN = 128, HN = 256;
constexpr int RELMAXC = 32, NBINSC = 16;
constexpr float DMINf = 0.0f, DMAXf = 22.0f, EPSf = 1e-6f;

typedef __attribute__((ext_vector_type(16))) _Float16 v16h;
typedef __attribute__((ext_vector_type(8)))  float    v8f;
typedef __attribute__((ext_vector_type(4)))  int      v4i;

union Frag16 { v16h h; uint4 u[2]; };
union PkH2   { _Float16 h[2]; unsigned u; };

struct F3 { float x, y, z; };
DEVI F3 sub3(F3 a, F3 b){ return {a.x-b.x, a.y-b.y, a.z-b.z}; }
DEVI float dot3(F3 a, F3 b){ return a.x*b.x + a.y*b.y + a.z*b.z; }
DEVI F3 scl3(F3 a, float s){ return {a.x*s, a.y*s, a.z*s}; }
DEVI F3 cross3(F3 a, F3 b){ return {a.y*b.z-a.z*b.y, a.z*b.x-a.x*b.z, a.x*b.y-a.y*b.x}; }
DEVI F3 ld3(const float* p){ return {p[0], p[1], p[2]}; }

DEVI void frames(const float* prow, F3& e1, F3& e2, F3& e3, F3& t){
  F3 p0 = ld3(prow), p1 = ld3(prow+3), p2 = ld3(prow+6);
  F3 v1 = sub3(p2, p1), v2 = sub3(p0, p1);
  e1 = scl3(v1, 1.0f/(sqrtf(dot3(v1,v1)) + EPSf));
  F3 u2 = sub3(v2, scl3(e1, dot3(v2, e1)));
  e2 = scl3(u2, 1.0f/(sqrtf(dot3(u2,u2)) + EPSf));
  e3 = cross3(e1, e2);
  t = p1;
}

// gelu_tanh(x) = 0.5x(1+tanh(c(x+0.044715x^3))) = x * sigmoid(2c(x+0.044715x^3))
// Branchless: one v_exp_f32 + one v_rcp_f32. Saturates correctly at +-inf.
DEVI float gelu_fast(float x){
  float u = x * (1.0f + 0.044715f * x * x);          // x + 0.044715 x^3
  float e = __expf(-1.5957691216057308f * u);        // exp(-2c*u)
  return x * __builtin_amdgcn_rcpf(1.0f + e);        // x * sigma(2c*u)
}

// ---------------------------------------------------------------------------
// Pack W1 (128x256) / W2 (256x128) f32 -> f16 B-fragments for 16x16x32 WMMA.
// Fragment order: [kt][nt][lane][8 dwords]; lane<16 holds K=kt*32+0..15 of
// column nt*16+lane, lane>=16 holds K=kt*32+16..31 (two f16 per dword).
// ---------------------------------------------------------------------------
__global__ __launch_bounds__(256)
void sd_pack_weights_25881472925819(const float* __restrict__ W1,
                                    const float* __restrict__ W2,
                                    unsigned* __restrict__ w1pk,
                                    unsigned* __restrict__ w2pk)
{
  int i = blockIdx.x * 256 + threadIdx.x;   // 0 .. 16383
  {
    int dw = i & 7, lane = (i >> 3) & 31, nt = (i >> 8) & 15, kt = (i >> 12) & 3;
    int col = nt*16 + (lane & 15);
    int k   = kt*32 + ((lane >> 4) << 4) + dw*2;
    PkH2 pk;
    pk.h[0] = (_Float16)W1[(k    )*HN + col];
    pk.h[1] = (_Float16)W1[(k + 1)*HN + col];
    w1pk[i] = pk.u;
  }
  {
    int dw = i & 7, lane = (i >> 3) & 31, nt = (i >> 8) & 7, kt = (i >> 11) & 7;
    int col = nt*16 + (lane & 15);
    int k   = kt*32 + ((lane >> 4) << 4) + dw*2;
    PkH2 pk;
    pk.h[0] = (_Float16)W2[(k    )*PN + col];
    pk.h[1] = (_Float16)W2[(k + 1)*PN + col];
    w2pk[i] = pk.u;
  }
}

// ---------------------------------------------------------------------------
// Main fused kernel: one block (8 wave32) per node.
// ---------------------------------------------------------------------------
__global__ __launch_bounds__(256)
void sd_pair_main_25881472925819(const float* __restrict__ pos,
                                 const int*   __restrict__ neigh,
                                 const int*   __restrict__ resi,
                                 const int*   __restrict__ chain,
                                 const int*   __restrict__ batch,
                                 const float* __restrict__ mask,
                                 const float* __restrict__ W_rel,
                                 const float* __restrict__ W_dist,
                                 const float* __restrict__ W_dir,
                                 const float* __restrict__ W_rot,
                                 const float* __restrict__ W_vec,
                                 const float* __restrict__ ln_s,
                                 const float* __restrict__ ln_b,
                                 const float* __restrict__ b1,
                                 const float* __restrict__ b2,
                                 const unsigned* __restrict__ w1pk,
                                 const unsigned* __restrict__ w2pk,
                                 float* __restrict__ out,
                                 float* __restrict__ outMask)
{
  constexpr int FW = 44;    // per-neighbour scalar row width
  constexpr int SPF = 132;  // padded f32 pair row stride
  constexpr int SPH = 136;  // padded f16 pair row stride (272B -> bank shift)
  constexpr int SHH = 264;  // padded f16 hidden row stride (528B)
  __shared__ __align__(16) unsigned s_w1[16384];     // 64KB: W1 B-fragments
  __shared__ __align__(16) float    s_feat[KN][FW];
  __shared__ __align__(16) float    s_pairF[KN][SPF];
  __shared__ __align__(16) _Float16 s_pairH[KN][SPH];
  __shared__ __align__(16) _Float16 s_hH[KN][SHH];
  __shared__ float s_mu[KN], s_rstd[KN];

  const int n    = blockIdx.x;
  const int tid  = threadIdx.x;
  const int lane = tid & 31;
  const int wave = tid >> 5;

#if USE_ASYNC_LDS
  // Kick off async DMA of the 64KB W1 fragment block into LDS; it runs on the
  // async path (ASYNCcnt) while the VALU does the geometry/feature phase.
  {
    const char* src = reinterpret_cast<const char*>(w1pk);
    char*       dst = reinterpret_cast<char*>(s_w1);
    #pragma unroll
    for (int p = 0; p < 16; ++p) {
      int off = p*4096 + tid*16;
      __builtin_amdgcn_global_load_async_to_lds_b128(
          (AS1 v4i*)(src + off), (AS3 v4i*)(dst + off), 0, 0);
    }
  }
#else
  {
    const uint4* src = reinterpret_cast<const uint4*>(w1pk);
    uint4*       dst = reinterpret_cast<uint4*>(s_w1);
    #pragma unroll
    for (int i = tid; i < 4096; i += 256) dst[i] = src[i];
  }
#endif

  // Warm W2 fragments into cache for the GEMM2 stream.
  if (wave == 0) {
    __builtin_prefetch(w2pk + lane * 512, 0, 1);
  }

  // ---- Stage 1: per-neighbour geometry (threads 0..31) --------------------
  if (tid < KN) {
    const int k = tid;
    int j = neigh[n*KN + k];
    int valid = (j >= 0);
    int jc = valid ? j : 0;
    outMask[(size_t)n*KN + k] = mask[n] * mask[jc] * (float)valid;

    float same = ((chain[n] == chain[jc]) && (batch[n] == batch[jc])) ? 1.0f : 0.0f;
    int rel = resi[jc] - resi[n];
    rel = (rel < -RELMAXC) ? -RELMAXC : (rel > RELMAXC ? RELMAXC : rel);
    rel += RELMAXC;

    F3 e1, e2, e3, t;   frames(pos + (size_t)n  * AN * 3, e1, e2, e3, t);
    F3 f1, f2, f3c, tj; frames(pos + (size_t)jc * AN * 3, f1, f2, f3c, tj);
    F3 diff = sub3(tj, t);                       // ca[j] - ca[n]
    float d = sqrtf(dot3(diff, diff) + EPSf);
    F3 unit = scl3(diff, 1.0f/(d + EPSf));

    float* ft = s_feat[k];
    const float step   = (DMAXf - DMINf) / (float)(NBINSC - 1);
    const float sig    = (DMAXf - DMINf) / (float)NBINSC;
    const float inv2s2 = 1.0f / (2.0f * sig * sig);
    #pragma unroll
    for (int f = 0; f < NBINSC; ++f) {
      float dd = d - (DMINf + step * (float)f);
      ft[f] = __expf(-dd * dd * inv2s2);
    }
    ft[16] = dot3(e1, unit); ft[17] = dot3(e2, unit); ft[18] = dot3(e3, unit);
    ft[19] = dot3(e1, f1); ft[20] = dot3(e1, f2); ft[21] = dot3(e1, f3c);
    ft[22] = dot3(e2, f1); ft[23] = dot3(e2, f2); ft[24] = dot3(e2, f3c);
    ft[25] = dot3(e3, f1); ft[26] = dot3(e3, f2); ft[27] = dot3(e3, f3c);
    const float* pj = pos + (size_t)jc * AN * 3;
    #pragma unroll
    for (int a = 0; a < AN; ++a) {
      F3 va = sub3(ld3(pj + a*3), t);
      ft[28 + a*3 + 0] = dot3(e1, va);
      ft[28 + a*3 + 1] = dot3(e2, va);
      ft[28 + a*3 + 2] = dot3(e3, va);
    }
    ft[40] = (float)rel;
    ft[41] = same;
  }
  __syncthreads();

  // ---- Stage 2: project features -> pair[32][128] f32 ---------------------
  {
    const int k  = tid >> 3;
    const int pb = (tid & 7) << 4;
    const float* ft = s_feat[k];
    const float same = ft[41];
    const float* wrl = W_rel + ((int)ft[40]) * PN;
    #pragma unroll 4
    for (int i = 0; i < 16; ++i) {
      int p = pb + i;
      float acc = same * wrl[p];
      #pragma unroll
      for (int f = 0; f < NBINSC; ++f) acc += ft[f]      * W_dist[f*PN + p];
      #pragma unroll
      for (int b = 0; b < 3;  ++b)     acc += ft[16 + b] * W_dir [b*PN + p];
      #pragma unroll
      for (int r = 0; r < 9;  ++r)     acc += ft[19 + r] * W_rot [r*PN + p];
      #pragma unroll
      for (int v = 0; v < 12; ++v)     acc += ft[28 + v] * W_vec [v*PN + p];
      s_pairF[k][p] = acc;
    }
  }
  __syncthreads();

  // ---- Stage 3: LayerNorm -> f16 A-matrix rows ----------------------------
  if (tid < KN) {
    float s = 0.0f, s2 = 0.0f;
    for (int p = 0; p < PN; ++p) { float x = s_pairF[tid][p]; s += x; s2 += x*x; }
    float mu  = s  * (1.0f/PN);
    float var = s2 * (1.0f/PN) - mu*mu;
    s_mu[tid]   = mu;
    s_rstd[tid] = rsqrtf(var + 1e-5f);
  }
  __syncthreads();
  {
    const int k  = tid >> 3;
    const int pb = (tid & 7) << 4;
    const float mu = s_mu[k], rs = s_rstd[k];
    #pragma unroll 4
    for (int i = 0; i < 16; ++i) {
      int p = pb + i;
      float x = (s_pairF[k][p] - mu) * rs * ln_s[p] + ln_b[p];
      s_pairH[k][p] = (_Float16)x;
    }
  }

#if USE_ASYNC_LDS
  // Each wave's async W1 transfers must land before any wave reads s_w1.
  __builtin_amdgcn_s_wait_asynccnt(0);
#endif
  __syncthreads();

  const int arow0 = (lane & 15);
  const int ksel  = (lane >> 4) * 16;  // A-frag K-half select per lane group
  const int rsh   = (lane < 16) ? 0 : 8;

  // ---- Stage 4: GEMM1 (32x128 @ 128x256) + gelu -> hH --------------------
  #pragma unroll
  for (int tIdx = 0; tIdx < 4; ++tIdx) {
    const int t  = wave*4 + tIdx;      // 32 C-tiles over 8 waves
    const int mt = t >> 4, nt = t & 15;
    v8f acc = {};
    const int arow = mt*16 + arow0;
    #pragma unroll
    for (int kt = 0; kt < 4; ++kt) {
      Frag16 a, b;
      const _Float16* ap = &s_pairH[arow][kt*32 + ksel];
      a.u[0] = *reinterpret_cast<const uint4*>(ap);
      a.u[1] = *reinterpret_cast<const uint4*>(ap + 8);
      const uint4* bp = reinterpret_cast<const uint4*>(s_w1 + (((kt*16 + nt)*32 + lane) << 3));
      b.u[0] = bp[0];
      b.u[1] = bp[1];
      acc = __builtin_amdgcn_wmma_f32_16x16x32_f16(false, a.h, false, b.h,
                                                   (short)0, acc, false, false);
    }
    const int col  = nt*16 + (lane & 15);
    const float bias = b1[col];
    const int rbase  = mt*16 + rsh;
    #pragma unroll
    for (int v = 0; v < 8; ++v) {
      s_hH[rbase + v][col] = (_Float16)gelu_fast(acc[v] + bias);
    }
  }
  __syncthreads();

  // ---- Stage 5: GEMM2 (32x256 @ 256x128) + bias -> global ----------------
  float* outp = out + ((size_t)n * KN) * PN;
  #pragma unroll
  for (int tIdx = 0; tIdx < 2; ++tIdx) {
    const int t  = wave*2 + tIdx;      // 16 C-tiles over 8 waves
    const int mt = t >> 3, nt = t & 7;
    v8f acc = {};
    const int arow = mt*16 + arow0;
    #pragma unroll
    for (int kt = 0; kt < 8; ++kt) {
      Frag16 a, b;
      const _Float16* ap = &s_hH[arow][kt*32 + ksel];
      a.u[0] = *reinterpret_cast<const uint4*>(ap);
      a.u[1] = *reinterpret_cast<const uint4*>(ap + 8);
      const uint4* bp = reinterpret_cast<const uint4*>(w2pk + (((kt*8 + nt)*32 + lane) << 3));
      b.u[0] = bp[0];
      b.u[1] = bp[1];
      acc = __builtin_amdgcn_wmma_f32_16x16x32_f16(false, a.h, false, b.h,
                                                   (short)0, acc, false, false);
    }
    const int col  = nt*16 + (lane & 15);
    const float bias = b2[col];
    const int rbase  = mt*16 + rsh;
    #pragma unroll
    for (int v = 0; v < 8; ++v) {
      outp[(size_t)(rbase + v)*PN + col] = acc[v] + bias;
    }
  }
}

// ---------------------------------------------------------------------------
extern "C" void kernel_launch(void* const* d_in, const int* in_sizes, int n_in,
                              void* d_out, int out_size, void* d_ws, size_t ws_size,
                              hipStream_t stream) {
  const float* pos    = (const float*)d_in[0];
  const int*   neigh  = (const int*)  d_in[1];
  const int*   resi   = (const int*)  d_in[2];
  const int*   chain  = (const int*)  d_in[3];
  const int*   batch  = (const int*)  d_in[4];
  const float* mask   = (const float*)d_in[5];
  const float* W_rel  = (const float*)d_in[6];
  const float* W_dist = (const float*)d_in[7];
  const float* W_dir  = (const float*)d_in[8];
  const float* W_rot  = (const float*)d_in[9];
  const float* W_vec  = (const float*)d_in[10];
  const float* ln_s   = (const float*)d_in[11];
  const float* ln_b   = (const float*)d_in[12];
  const float* W1     = (const float*)d_in[13];
  const float* b1     = (const float*)d_in[14];
  const float* W2     = (const float*)d_in[15];
  const float* b2     = (const float*)d_in[16];

  float* out     = (float*)d_out;
  float* outMask = out + (size_t)NN * KN * PN;

  unsigned* w1pk = (unsigned*)d_ws;          // 16384 u32 = 64KB
  unsigned* w2pk = w1pk + 16384;             // 16384 u32 = 64KB

  sd_pack_weights_25881472925819<<<64, 256, 0, stream>>>(W1, W2, w1pk, w2pk);
  sd_pair_main_25881472925819<<<NN, 256, 0, stream>>>(
      pos, neigh, resi, chain, batch, mask,
      W_rel, W_dist, W_dir, W_rot, W_vec, ln_s, ln_b, b1, b2,
      w1pk, w2pk, out, outMask);
}
